// Geoneigh_att_37855841747279
// MI455X (gfx1250) — matrix operand order repaired and checked by reading, not compiled, not run on hardware
//
#include <hip/hip_runtime.h>

typedef __attribute__((ext_vector_type(16))) _Float16 v16h;
typedef __attribute__((ext_vector_type(8)))  float    v8f;
typedef __attribute__((ext_vector_type(4)))  _Float16 h4;
typedef __attribute__((ext_vector_type(4)))  int      v4i;

#define BB   2
#define NN   4096
#define NS   32
#define NH   8
#define HD   32
#define GG   16
#define DD   48                 // HD + G
#define NC   (NH * DD)          // 384 "columns" of the per-position GEMM
#define SCALE 0.17677669529663687f   // 32^-0.5

#define AS1 __attribute__((address_space(1)))
#define AS3 __attribute__((address_space(3)))

#if defined(__has_builtin)
# if __has_builtin(__builtin_amdgcn_global_load_async_to_lds_b128)
#  define HAVE_ASYNC_COPY 1
# endif
# if __has_builtin(__builtin_amdgcn_s_wait_asynccnt)
#  define HAVE_WAIT_ASYNC 1
# endif
#endif

__global__ __launch_bounds__(256) void geo_att_kernel(
    const float* __restrict__ q,
    const float* __restrict__ kn,
    const float* __restrict__ vn,
    const float* __restrict__ geo,
    const float* __restrict__ W1,
    const float* __restrict__ b1,
    const float* __restrict__ W2,
    const float* __restrict__ b2,
    float* __restrict__ out)
{
    __shared__ _Float16 s_we[NS * NC];      // 24 KB: we[s][c], c = h*48 + d
    __shared__ float    s_v[NS * NH * HD];  // 32 KB: v staged via async-to-LDS
    __shared__ float    s_q[NH * HD];       // 1 KB
    __shared__ float    s_w1[NS * NS];      // 4 KB
    __shared__ float    s_w2[NS];
    __shared__ float    s_score[NC];        // 1.5 KB
    __shared__ float    s_wt[NS * NH];      // weight -> attn, in place
    __shared__ float    s_const[1];

    const int tid    = threadIdx.x;
    const int lane   = tid & 31;
    const int wave   = tid >> 5;
    const int nIdx   = lane & 15;           // M for A, N for B/C
    const int halfId = lane >> 4;

    const long pos   = blockIdx.x;                   // b*N + n
    const long kbase = pos * (long)(NS * NH * HD);   // 8192 elems / position
    const long gbase = pos * (long)(NS * GG);
    const long qbase = pos * (long)(NH * HD);

    // ---------- phase A: launch async v_neigh -> LDS (overlaps everything) ----
#if defined(HAVE_ASYNC_COPY)
    #pragma unroll
    for (int j = 0; j < 8; ++j) {
        const int i4 = tid + j * 256;                // float4 index, 2048 total
        __builtin_amdgcn_global_load_async_to_lds_b128(
            (AS1 v4i*)(vn + kbase + 4 * (long)i4),
            (AS3 v4i*)&s_v[4 * i4], 0, 0);
    }
#endif

    // ---------- phase 0: stage shared operands ----------
    if (tid < NH * HD) s_q[tid] = q[qbase + tid];
    for (int i = tid; i < NS * NS; i += 256) s_w1[i] = W1[i];
    if (tid < NS) s_w2[tid] = W2[tid];
    if (tid == 0) {
        float sc = b2[0];                            // score bias: b2 + W2 . b1
        for (int t = 0; t < NS; ++t) sc += W2[t] * b1[t];
        s_const[0] = sc;
    }
    __syncthreads();

    // ---------- phase 1: build we (f16) in LDS, init score ----------
    const float cinit = s_const[0];
    for (int i = tid; i < NC; i += 256) s_score[i] = cinit;

#if !defined(HAVE_ASYNC_COPY)
    #pragma unroll
    for (int j = 0; j < 8; ++j) {
        const int i4 = tid + j * 256;
        *(float4*)&s_v[4 * i4] = *(const float4*)&vn[kbase + 4 * (long)i4];
    }
#endif

    // subtract-kernel part, b128 global loads: 2048 float4 / block
    #pragma unroll
    for (int j = 0; j < 8; ++j) {
        const int i4 = tid + j * 256;
        const int s  = i4 >> 6;
        const int f4 = i4 & 63;                      // float4 within (h,d) plane
        const int h  = f4 >> 3;
        const int d4 = (f4 & 7) << 2;
        const float4 kv = *(const float4*)&kn[kbase + 4 * (long)i4];
        const float4 qv = *(const float4*)&s_q[f4 * 4];
        h4 wv;
        wv[0] = (_Float16)(qv.x * SCALE - kv.x);
        wv[1] = (_Float16)(qv.y * SCALE - kv.y);
        wv[2] = (_Float16)(qv.z * SCALE - kv.z);
        wv[3] = (_Float16)(qv.w * SCALE - kv.w);
        *(h4*)&s_we[s * NC + h * DD + d4] = wv;      // 8B-aligned ds_store_b64
    }
    // geometric features broadcast over heads (128 float4 total)
    if (tid < 128) {
        const int s  = tid >> 2;
        const int g4 = (tid & 3) << 2;
        const float4 gv = *(const float4*)&geo[gbase + (long)tid * 4];
        h4 hv;
        hv[0] = (_Float16)gv.x; hv[1] = (_Float16)gv.y;
        hv[2] = (_Float16)gv.z; hv[3] = (_Float16)gv.w;
        #pragma unroll
        for (int h = 0; h < NH; ++h)
            *(h4*)&s_we[s * NC + h * DD + HD + g4] = hv;
    }
    __syncthreads();

    // ---------- phase 2: H1 = W1 x WE on the matrix pipe, fold W2 ----------
    // A fragments (16x32 f16, M x K): K(e,half) = (e&7) + 8*half + 16*(e>>3)
    v16h a0, a1;
    #pragma unroll
    for (int e = 0; e < 16; ++e) {
        const int K = (e & 7) + 8 * halfId + ((e >> 3) << 4);
        a0[e] = (_Float16)s_w1[(nIdx)      * NS + K];   // t = 0..15
        a1[e] = (_Float16)s_w1[(16 + nIdx) * NS + K];   // t = 16..31
    }

    #pragma unroll
    for (int j = 0; j < 3; ++j) {
        const int ct = wave * 3 + j;         // column tile 0..23 (uniform per wave)
        // B fragment (32x16 f16, K x N): K(e,half) = (e&7) + 8*half + 16*(e>>3)
        v16h bf;
        #pragma unroll
        for (int e = 0; e < 16; ++e) {
            const int K = (e & 7) + 8 * halfId + ((e >> 3) << 4);
            bf[e] = s_we[K * NC + ct * 16 + nIdx];
        }
        v8f c0 = {}, c1 = {};
        c0 = __builtin_amdgcn_wmma_f32_16x16x32_f16(false, a0, false, bf,
                                                    (short)0, c0, false, false);
        c1 = __builtin_amdgcn_wmma_f32_16x16x32_f16(false, a1, false, bf,
                                                    (short)0, c1, false, false);
        // fold W2 over t: C VGPR k -> t = 8*halfId + k (c0), 16 + 8*halfId + k (c1)
        float p = 0.f;
        #pragma unroll
        for (int k = 0; k < 8; ++k) {
            p += s_w2[8 * halfId + k]      * c0[k];
            p += s_w2[16 + 8 * halfId + k] * c1[k];
        }
        atomicAdd(&s_score[ct * 16 + nIdx], p);   // ds_add_f32: combines lane halves
    }
    __syncthreads();

    // ---------- phase 3: weight[s][h] = sum_d score[h,d] * we[s][h,d] ----------
    {
        const int s = tid >> 3, h = tid & 7;
        float acc = 0.f;
        #pragma unroll
        for (int c4 = 0; c4 < DD / 4; ++c4) {
            const h4 wv = *(const h4*)&s_we[s * NC + h * DD + 4 * c4];
            const float* sr = &s_score[h * DD + 4 * c4];
            acc += (float)wv[0] * sr[0] + (float)wv[1] * sr[1]
                 + (float)wv[2] * sr[2] + (float)wv[3] * sr[3];
        }
        s_wt[s * NH + h] = acc;
    }
    __syncthreads();

    // ---------- phase 4: softmax over neighbors (per head) ----------
    if (tid < NH) {
        const int h = tid;
        float m = -3.4e38f;
        for (int s = 0; s < NS; ++s) m = fmaxf(m, s_wt[s * NH + h]);
        float sum = 0.f;
        for (int s = 0; s < NS; ++s) {
            const float e = __expf(s_wt[s * NH + h] - m);
            s_wt[s * NH + h] = e;
            sum += e;
        }
        const float inv = 1.f / sum;
        for (int s = 0; s < NS; ++s) s_wt[s * NH + h] *= inv;
    }

    // all async v tiles must be resident before the post-barrier LDS reads
#if defined(HAVE_ASYNC_COPY)
# if defined(HAVE_WAIT_ASYNC)
    __builtin_amdgcn_s_wait_asynccnt(0);
# else
    asm volatile("s_wait_asynccnt 0" ::: "memory");
# endif
#endif
    __syncthreads();

    // ---------- phase 5: out[h,d] = sum_s attn[s,h] * v[s,h,d] (from LDS) -----
    {
        const int h = tid >> 5;
        float acc = 0.f;
        #pragma unroll 8
        for (int s = 0; s < NS; ++s)
            acc += s_wt[s * NH + h] * s_v[s * (NH * HD) + tid];
        out[qbase + tid] = acc;
    }
}

extern "C" void kernel_launch(void* const* d_in, const int* in_sizes, int n_in,
                              void* d_out, int out_size, void* d_ws, size_t ws_size,
                              hipStream_t stream) {
    const float* q   = (const float*)d_in[0];
    const float* kn  = (const float*)d_in[1];
    const float* vn  = (const float*)d_in[2];
    const float* geo = (const float*)d_in[3];
    const float* W1  = (const float*)d_in[4];
    const float* b1  = (const float*)d_in[5];
    const float* W2  = (const float*)d_in[6];
    const float* b2  = (const float*)d_in[7];
    float* out = (float*)d_out;
    (void)in_sizes; (void)n_in; (void)out_size; (void)d_ws; (void)ws_size;

    geo_att_kernel<<<BB * NN, 256, 0, stream>>>(q, kn, vn, geo, W1, b1, W2, b2, out);
}
